// MoE_ffn_1357209665613
// MI455X (gfx1250) — matrix operand order repaired and checked by reading, not compiled, not run on hardware
//
#include <hip/hip_runtime.h>
#include <hip/hip_bf16.h>

// Problem constants (from reference): N=32768, D=768, E=8, F=4D=3072, top_k=2
#define N_TOK 32768
#define DDIM  768
#define FDIM  3072
#define NEXP  8
#define TOPK  2
#define TM    32      // tokens per workgroup in main kernel (2 row tiles)
#define NB    4       // column tiles per wave block (4 col tiles)
#define WAVES 8       // 256 threads = 8 wave32

typedef _Float16 v16h __attribute__((ext_vector_type(16)));
typedef _Float16 v8h  __attribute__((ext_vector_type(8)));
typedef float    v8f  __attribute__((ext_vector_type(8)));

union ABFrag { v16h v; v8h h[2]; };

// ---------------------------------------------------------------------------
// 1) Convert first two experts of W1 (F x D each, row-major) to f16
// ---------------------------------------------------------------------------
__global__ void k_convert_w1(const float* __restrict__ W1,
                             _Float16* __restrict__ w1h, int n) {
    int i = blockIdx.x * blockDim.x + threadIdx.x;
    int stride = gridDim.x * blockDim.x;
    for (; i < n; i += stride) w1h[i] = (_Float16)W1[i];
}

// ---------------------------------------------------------------------------
// 2a) v_k[f] = sum_d W2[k,d,f] * W3[k,0,d]   (collapses second GEMM)
// ---------------------------------------------------------------------------
__global__ void k_compute_v(const float* __restrict__ W2,
                            const float* __restrict__ W3,
                            float* __restrict__ vvec) {
    int idx = blockIdx.x * blockDim.x + threadIdx.x;
    if (idx >= TOPK * FDIM) return;
    int k = idx / FDIM, f = idx % FDIM;
    const float* w2 = W2 + (size_t)k * DDIM * FDIM + f;
    const float* w3 = W3 + (size_t)k * DDIM;
    float s = 0.f;
    for (int d = 0; d < DDIM; ++d) s += w2[(size_t)d * FDIM] * w3[d];
    vvec[idx] = s;
}

// ---------------------------------------------------------------------------
// 2b) c_k = b3[k] + sum_d b2[k,d] * W3[k,0,d]     (one wave per expert)
// ---------------------------------------------------------------------------
__global__ void k_compute_c(const float* __restrict__ b2,
                            const float* __restrict__ b3,
                            const float* __restrict__ W3,
                            float* __restrict__ cvec) {
    int k = threadIdx.x >> 5, lane = threadIdx.x & 31;
    float s = 0.f;
    for (int d = lane; d < DDIM; d += 32)
        s += b2[k * DDIM + d] * W3[k * DDIM + d];
    #pragma unroll
    for (int m = 16; m >= 1; m >>= 1) s += __shfl_xor(s, m, 32);
    if (lane == 0) cvec[k] = s + b3[k];
}

// ---------------------------------------------------------------------------
// 3) Gating + residual scalars: one wave per token.
//    gate[n] = { w0, w1, t0, t1 }  with  w0 = 1/(1+exp(l2-l1)),  t_k = x[n]·W3_k
// ---------------------------------------------------------------------------
__global__ void k_gate(const float* __restrict__ x,
                       const float* __restrict__ Wg,
                       const float* __restrict__ W3,
                       float* __restrict__ gate) {
    const int wave  = threadIdx.x >> 5;
    const int lane  = threadIdx.x & 31;
    const int token = blockIdx.x * (blockDim.x >> 5) + wave;
    if (token >= N_TOK) return;

    float acc[NEXP + 2];
    #pragma unroll
    for (int i = 0; i < NEXP + 2; ++i) acc[i] = 0.f;

    const float* xr = x + (size_t)token * DDIM;
    for (int d = lane; d < DDIM; d += 32) {
        float xv = xr[d];
        #pragma unroll
        for (int e = 0; e < NEXP; ++e) acc[e] += xv * Wg[e * DDIM + d];
        acc[NEXP + 0] += xv * W3[d];            // expert 0 residual dot
        acc[NEXP + 1] += xv * W3[DDIM + d];     // expert 1 residual dot
    }
    #pragma unroll
    for (int m = 16; m >= 1; m >>= 1) {
        #pragma unroll
        for (int i = 0; i < NEXP + 2; ++i) acc[i] += __shfl_xor(acc[i], m, 32);
    }
    if (lane == 0) {
        float l1 = -3.4e38f, l2 = -3.4e38f;
        #pragma unroll
        for (int e = 0; e < NEXP; ++e) {
            float le = acc[e];
            if (le > l1) { l2 = l1; l1 = le; }
            else if (le > l2) { l2 = le; }
        }
        float w0 = 1.f / (1.f + __expf(l2 - l1));
        gate[(size_t)token * 4 + 0] = w0;
        gate[(size_t)token * 4 + 1] = 1.f - w0;
        gate[(size_t)token * 4 + 2] = acc[NEXP + 0];
        gate[(size_t)token * 4 + 3] = acc[NEXP + 1];
    }
}

// ---------------------------------------------------------------------------
// 4) Main fused kernel: H = relu(X @ W1_k^T + b1_k), s_k[n] = H·v_k,
//    out[n] = w0*(t0+c0+s0) + w1*(t1+c1+s1)
//    32 tokens / WG; each wave owns a 2x4 block of 16x16 accumulator tiles:
//    A fragments feed 4 wmma each (LDS traffic /4), B fragments feed 2 each.
// ---------------------------------------------------------------------------
__launch_bounds__(256)
__global__ void k_moe_main(const float* __restrict__ x,
                           const _Float16* __restrict__ w1h,
                           const float* __restrict__ b1,
                           const float* __restrict__ vvec,
                           const float* __restrict__ cvec,
                           const float* __restrict__ gate,
                           float* __restrict__ out) {
    __shared__ _Float16 xs[TM * DDIM];   // 32x768 f16 = 48 KB
    __shared__ float    sred[TOPK][TM];

    const int tid  = threadIdx.x;
    const int lane = tid & 31;
    const int wave = tid >> 5;
    const int n0   = blockIdx.x * TM;

    // Stage + convert the token tile into LDS (coalesced f32 reads).
    for (int i = tid; i < TM * DDIM; i += 256)
        xs[i] = (_Float16)x[(size_t)n0 * DDIM + i];
    if (tid < TOPK * TM) (&sred[0][0])[tid] = 0.f;
    __syncthreads();

    // CDNA5 16-bit WMMA fragment addressing (wave32):
    //  A (16x32, M=lane%16):  halves [0..7] = K + (lane<16?0:8), [8..15] = +16
    //  B (32x16, N=lane%16):  halves [0..7] = K + (lane<16?0:8), [8..15] = +16
    const int col  = lane & 15;
    const int asel = (lane < 16) ? 0 : 8;    // in halves
    const int bsel = (lane < 16) ? 0 : 8;    // in halves

    const _Float16* __restrict__ arow0 = xs + col * DDIM + asel;          // tokens 0..15
    const _Float16* __restrict__ arow1 = xs + (col + 16) * DDIM + asel;   // tokens 16..31

    for (int k = 0; k < TOPK; ++k) {
        const _Float16* __restrict__ wk = w1h + (size_t)k * FDIM * DDIM;
        float sp0[8], sp1[8];
        #pragma unroll
        for (int r = 0; r < 8; ++r) { sp0[r] = 0.f; sp1[r] = 0.f; }

        // 192 column tiles; each wave-iteration covers NB=4 consecutive tiles.
        for (int g = wave; g < FDIM / (16 * NB); g += WAVES) {
            const int f0 = g * (16 * NB);
            // One B base pointer; the 4 column slabs sit at immediate
            // offsets J * 16*DDIM*2 = J * 24576 bytes (fits 24-bit ioffset).
            const _Float16* __restrict__ brow = wk + (size_t)(f0 + col) * DDIM + bsel;
            // Prefetch next group's B slabs (global_prefetch_b8).
            #pragma unroll
            for (int J = 0; J < NB; ++J)
                __builtin_prefetch(brow + ((size_t)16 * NB * WAVES + 16 * J) * DDIM, 0, 1);

            v8f acc[2][NB];
            #pragma unroll
            for (int J = 0; J < NB; ++J) { acc[0][J] = (v8f){}; acc[1][J] = (v8f){}; }

            #pragma unroll 2
            for (int kk = 0; kk < DDIM; kk += 32) {
                ABFrag a0, a1;
                a0.h[0] = *(const v8h*)(arow0 + kk);
                a0.h[1] = *(const v8h*)(arow0 + kk + 16);
                a1.h[0] = *(const v8h*)(arow1 + kk);
                a1.h[1] = *(const v8h*)(arow1 + kk + 16);
                #pragma unroll
                for (int J = 0; J < NB; ++J) {
                    ABFrag b;
                    b.h[0] = *(const v8h*)(brow + (size_t)J * 16 * DDIM + kk);
                    b.h[1] = *(const v8h*)(brow + (size_t)J * 16 * DDIM + kk + 16);
                    acc[0][J] = __builtin_amdgcn_wmma_f32_16x16x32_f16(
                        false, a0.v, false, b.v, (short)0, acc[0][J], false, false);
                    acc[1][J] = __builtin_amdgcn_wmma_f32_16x16x32_f16(
                        false, a1.v, false, b.v, (short)0, acc[1][J], false, false);
                }
            }
            // Fused epilogue: relu(+b1) then dot with v_k along F.
            #pragma unroll
            for (int J = 0; J < NB; ++J) {
                const float bvf = b1[k * FDIM + f0 + 16 * J + col];
                const float vvf = vvec[k * FDIM + f0 + 16 * J + col];
                #pragma unroll
                for (int r = 0; r < 8; ++r) {
                    float h0 = acc[0][J][r] + bvf; h0 = h0 > 0.f ? h0 : 0.f;
                    float h1 = acc[1][J][r] + bvf; h1 = h1 > 0.f ? h1 : 0.f;
                    sp0[r] += h0 * vvf;
                    sp1[r] += h1 * vvf;
                }
            }
        }
        // Reduce over the 16 columns held in each lane-half.
        #pragma unroll
        for (int m = 8; m >= 1; m >>= 1) {
            #pragma unroll
            for (int r = 0; r < 8; ++r) {
                sp0[r] += __shfl_xor(sp0[r], m, 16);
                sp1[r] += __shfl_xor(sp1[r], m, 16);
            }
        }
        // C/D layout: vgpr r holds row r (lanes 0-15) or row r+8 (lanes 16-31).
        if (lane == 0) {
            #pragma unroll
            for (int r = 0; r < 8; ++r) {
                atomicAdd(&sred[k][r],      sp0[r]);
                atomicAdd(&sred[k][16 + r], sp1[r]);
            }
        } else if (lane == 16) {
            #pragma unroll
            for (int r = 0; r < 8; ++r) {
                atomicAdd(&sred[k][8 + r],  sp0[r]);
                atomicAdd(&sred[k][24 + r], sp1[r]);
            }
        }
    }
    __syncthreads();

    if (tid < TM) {
        const int n = n0 + tid;
        const float w0 = gate[(size_t)n * 4 + 0];
        const float w1 = gate[(size_t)n * 4 + 1];
        const float t0 = gate[(size_t)n * 4 + 2];
        const float t1 = gate[(size_t)n * 4 + 3];
        out[n] = w0 * (t0 + cvec[0] + sred[0][tid]) +
                 w1 * (t1 + cvec[1] + sred[1][tid]);
    }
}

// ---------------------------------------------------------------------------
extern "C" void kernel_launch(void* const* d_in, const int* in_sizes, int n_in,
                              void* d_out, int out_size, void* d_ws, size_t ws_size,
                              hipStream_t stream) {
    (void)in_sizes; (void)n_in; (void)out_size; (void)ws_size;

    const float* x  = (const float*)d_in[0];  // (N, D)
    const float* Wg = (const float*)d_in[1];  // (E, D)
    const float* W1 = (const float*)d_in[2];  // (E, F, D)
    const float* b1 = (const float*)d_in[3];  // (E, F)
    const float* W2 = (const float*)d_in[4];  // (E, D, F)
    const float* b2 = (const float*)d_in[5];  // (E, D)
    const float* W3 = (const float*)d_in[6];  // (E, 1, D)
    const float* b3 = (const float*)d_in[7];  // (E, 1)
    float* out = (float*)d_out;               // (N, 1) f32

    // Workspace layout
    char* ws = (char*)d_ws;
    size_t off = 0;
    _Float16* w1h = (_Float16*)(ws + off);
    off += (size_t)TOPK * FDIM * DDIM * sizeof(_Float16);   // 9.44 MB
    off = (off + 255) & ~(size_t)255;
    float* vvec = (float*)(ws + off);
    off += (size_t)TOPK * FDIM * sizeof(float);             // 24 KB
    off = (off + 255) & ~(size_t)255;
    float* cvec = (float*)(ws + off);
    off += 256;
    float* gvec = (float*)(ws + off);                        // N*4 floats = 512 KB

    k_convert_w1<<<2048, 256, 0, stream>>>(W1, w1h, TOPK * FDIM * DDIM);
    k_compute_v<<<(TOPK * FDIM + 255) / 256, 256, 0, stream>>>(W2, W3, vvec);
    k_compute_c<<<1, 64, 0, stream>>>(b2, b3, W3, cvec);
    k_gate<<<N_TOK / (256 / 32), 256, 0, stream>>>(x, Wg, W3, gvec);
    k_moe_main<<<N_TOK / TM, 256, 0, stream>>>(x, w1h, b1, vvec, cvec, gvec, out);
}
